// KeypointEncoder_51591147160051
// MI455X (gfx1250) — compile-verified, matrix operand order checked by reading, not run on hardware
//
#include <hip/hip_runtime.h>
#include <hip/hip_bf16.h>
#include <cmath>

// ---- CDNA5 WMMA vector types ----
typedef __attribute__((ext_vector_type(16))) _Float16 v16h;
typedef __attribute__((ext_vector_type(8)))  _Float16 v8h;
typedef __attribute__((ext_vector_type(8)))  float    v8f;
typedef __attribute__((ext_vector_type(4)))  int      v4i;

// ---- problem constants ----
#define BB 32
#define NN 8192
#define MT 128            // points per workgroup
#define XS 264            // f16 activation LDS row stride (max Cin 256 + pad)
#define YS 260            // f32 GEMM-out LDS row stride (max Cout 256 + pad)
#define WBUF_HALVES 32768 // 64 KB LDS staging buffer for one weight chunk

// packed-f16 weight offsets (in halves) inside d_ws:
#define WOFF0 0
#define WOFF1 1024
#define WOFF2 3072
#define WOFF3 11264
#define WOFF4 44032

#if __has_builtin(__builtin_amdgcn_global_load_async_to_lds_b128)
#define HAS_ASYNC_LDS 1
#else
#define HAS_ASYNC_LDS 0
#endif

struct Params {
    const float* kpts;       // [B,2,N]
    const float* scores;     // [B,1,N]
    const _Float16* w;       // packed f16 weights in d_ws
    const float* bias[5];
    const float* gamma[4];
    const float* beta[4];
    float* out;              // [B,256,N]
};

// Load one 16-half WMMA fragment slice (two b128 loads).
// ISA 16-bit A/B layout: lane<16 holds K {base..+7, base+16..+23},
// lane>=16 holds K {base+8..+15, base+24..+31}; +8 lane offset folded by caller.
__device__ __forceinline__ v16h load_frag(const _Float16* base) {
    union { v16h v; v8h h[2]; } u;
    u.h[0] = *(const v8h*)(base);
    u.h[1] = *(const v8h*)(base + 16);
    return u.v;
}

// Issue async VMEM->LDS copies of a weight chunk (ASYNCcnt-tracked on gfx1250).
__device__ __forceinline__ void stage_issue(const _Float16* __restrict__ src,
                                            _Float16* dst, int halves, int tid) {
    const int nvec = halves >> 3;                    // 16B transfers
    for (int j = tid; j < nvec; j += 256) {
#if HAS_ASYNC_LDS
        __builtin_amdgcn_global_load_async_to_lds_b128(
            (__attribute__((address_space(1))) v4i*)(const_cast<_Float16*>(src) + (size_t)j * 8),
            (__attribute__((address_space(3))) v4i*)(dst + (size_t)j * 8),
            0, 0);
#else
        *(v8h*)(dst + (size_t)j * 8) = *(const v8h*)(src + (size_t)j * 8);
#endif
    }
}

__device__ __forceinline__ void stage_wait() {
#if HAS_ASYNC_LDS
#if __has_builtin(__builtin_amdgcn_s_wait_asynccnt)
    __builtin_amdgcn_s_wait_asynccnt(0);
#else
    asm volatile("s_wait_asynccnt 0" ::: "memory");
#endif
#endif
    __syncthreads();
}

// One fused layer: stage W chunk(s) to LDS (async), WMMA GEMM + bias, then
// channel-LayerNorm + exact GELU back into the f16 activation buffer, or the
// final coalesced store to HBM.
template<int CIN, int COUT, bool LAST>
__device__ void do_layer(const _Float16* __restrict__ Wl,   // [COUT][CIN] f16 row-major
                         const float* __restrict__ bs,
                         const float* __restrict__ g,
                         const float* __restrict__ be,
                         _Float16* xs, float* ys, float* red, _Float16* wbuf,
                         float* outp, int tid)
{
    const int lane = tid & 31;
    const int wave = tid >> 5;          // 8 waves; wave == its 16-row M tile
    const int lm   = lane & 15;
    const int hof  = (lane >> 4) << 3;  // +8 halves for lanes 16..31
    constexpr int NK       = CIN / 32;
    constexpr int TN_TOTAL = COUT / 16;
    constexpr int NCH      = (CIN * COUT * 2 > (WBUF_HALVES * 2)) ? 2 : 1;
    constexpr int TN_PER   = TN_TOTAL / NCH;
    constexpr int CH_HALVES = TN_PER * 16 * CIN;

    const int mbase = wave * 16 + ((lane >> 4) << 3);   // C/D: M = r + 8*(lane>=16)
    v16h a[NK];

    #pragma unroll
    for (int ch = 0; ch < NCH; ++ch) {
        // Kick off async weight staging, overlap with A-fragment hoist (ds loads).
        stage_issue(Wl + (size_t)ch * CH_HALVES, wbuf, CH_HALVES, tid);
        if (ch == 0) {
            const _Float16* rowA = xs + (wave * 16 + lm) * XS + hof;
            #pragma unroll
            for (int k = 0; k < NK; ++k) a[k] = load_frag(rowA + k * 32);
        }
        stage_wait();   // s_wait_asynccnt 0 + workgroup barrier

        for (int tl = 0; tl < TN_PER; ++tl) {
            const int tn = ch * TN_PER + tl;
            // B = W^T tile out of LDS: lane's output-channel row is contiguous in K.
            const _Float16* wrow = wbuf + (size_t)(tl * 16 + lm) * CIN + hof;
            v8f acc = {};
            #pragma unroll
            for (int k = 0; k < NK; ++k) {
                v16h b = load_frag(wrow + k * 32);
                acc = __builtin_amdgcn_wmma_f32_16x16x32_f16(
                    /*neg_a=*/false, a[k], /*neg_b=*/false, b,
                    /*c_mod=*/(short)0, acc, /*reuse_a=*/false, /*reuse_b=*/false);
            }
            const float bv  = bs[tn * 16 + lm];
            const int   col = tn * 16 + lm;
            #pragma unroll
            for (int r = 0; r < 8; ++r)
                ys[(mbase + r) * YS + col] = acc[r] + bv;
        }
        __syncthreads();  // wbuf consumed + ys chunk visible before restage/LN
    }

    if (!LAST) {
        // LayerNorm over channels + exact GELU; 2 threads per point.
        const int m  = tid & (MT - 1);
        const int h  = tid >> 7;
        const int cw = COUT / 2;
        const int c0 = h * cw;
        float s = 0.f, s2 = 0.f;
        for (int c = c0; c < c0 + cw; ++c) {
            float v = ys[m * YS + c];
            s += v; s2 += v * v;
        }
        red[m * 4 + h * 2 + 0] = s;
        red[m * 4 + h * 2 + 1] = s2;
        __syncthreads();
        const float S  = red[m * 4 + 0] + red[m * 4 + 2];
        const float Q  = red[m * 4 + 1] + red[m * 4 + 3];
        const float mu = S * (1.f / (float)COUT);
        const float var = Q * (1.f / (float)COUT) - mu * mu;
        const float rs = rsqrtf(var + 1e-5f);
        for (int c = c0; c < c0 + cw; ++c) {
            float v = (ys[m * YS + c] - mu) * rs;
            v = v * g[c] + be[c];
            float ge = 0.5f * v * (1.0f + erff(v * 0.70710678118654752f)); // exact GELU
            xs[m * XS + c] = (_Float16)ge;
        }
        __syncthreads();
    } else {
        // outp already points at &out[b*256*N + n0]; rows of 128 floats, coalesced.
        for (int idx = tid; idx < MT * 256; idx += 256) {
            int m = idx & (MT - 1);
            int c = idx >> 7;
            outp[(size_t)c * NN + m] = ys[m * YS + c];
        }
    }
}

__global__ __launch_bounds__(256)
void keypoint_encoder_kernel(Params p) {
    extern __shared__ char smem[];
    _Float16* xs   = (_Float16*)smem;                                       // [128][XS] f16
    float*    ys   = (float*)(smem + MT * XS * 2);                          // [128][YS] f32
    float*    red  = (float*)(smem + MT * XS * 2 + MT * YS * 4);            // [128][4]
    _Float16* wbuf = (_Float16*)(smem + MT * XS * 2 + MT * YS * 4 + MT * 16); // 64 KB

    const int tid = threadIdx.x;
    const long p0 = (long)blockIdx.x * MT;      // 64 blocks per batch image
    const int  b  = (int)(p0 / NN);
    const int  n0 = (int)(p0 % NN);

    if (tid == 0) __builtin_prefetch(p.w, 0, 0);   // global_prefetch_b8 (weights -> L2)

    // Stage input: [kpts_x, kpts_y, score, 0...0] padded to K=32 in f16.
    if (tid < MT) {
        const int n = n0 + tid;
        const float* kb = p.kpts + (size_t)b * 2 * NN;
        _Float16* row = xs + tid * XS;
        row[0] = (_Float16)kb[n];
        row[1] = (_Float16)kb[NN + n];
        row[2] = (_Float16)p.scores[(size_t)b * NN + n];
        #pragma unroll
        for (int c = 3; c < 32; ++c) row[c] = (_Float16)0.f;
    }
    __syncthreads();

    const _Float16* w = p.w;
    do_layer< 32,  32, false>(w + WOFF0, p.bias[0], p.gamma[0], p.beta[0], xs, ys, red, wbuf, nullptr, tid);
    do_layer< 32,  64, false>(w + WOFF1, p.bias[1], p.gamma[1], p.beta[1], xs, ys, red, wbuf, nullptr, tid);
    do_layer< 64, 128, false>(w + WOFF2, p.bias[2], p.gamma[2], p.beta[2], xs, ys, red, wbuf, nullptr, tid);
    do_layer<128, 256, false>(w + WOFF3, p.bias[3], p.gamma[3], p.beta[3], xs, ys, red, wbuf, nullptr, tid);
    float* outp = p.out + (size_t)b * 256 * NN + n0;
    do_layer<256, 256, true >(w + WOFF4, p.bias[4], nullptr,    nullptr,   xs, ys, red, wbuf, outp,    tid);
}

// Convert f32 weights -> f16, padding K (layer 0: 3 -> 32 with zeros).
__global__ void pack_weights_kernel(const float* __restrict__ src, _Float16* __restrict__ dst,
                                    int Cout, int CinSrc, int CinDst) {
    int i = blockIdx.x * blockDim.x + threadIdx.x;
    if (i >= Cout * CinDst) return;
    int o = i / CinDst, c = i % CinDst;
    dst[i] = (_Float16)((c < CinSrc) ? src[o * CinSrc + c] : 0.0f);
}

extern "C" void kernel_launch(void* const* d_in, const int* in_sizes, int n_in,
                              void* d_out, int out_size, void* d_ws, size_t ws_size,
                              hipStream_t stream) {
    (void)in_sizes; (void)n_in; (void)out_size; (void)ws_size;
    const float* kpts   = (const float*)d_in[0];
    const float* scores = (const float*)d_in[1];
    const float* W0 = (const float*)d_in[2];
    const float* b0 = (const float*)d_in[3];
    const float* g0 = (const float*)d_in[4];
    const float* e0 = (const float*)d_in[5];
    const float* W1 = (const float*)d_in[6];
    const float* b1 = (const float*)d_in[7];
    const float* g1 = (const float*)d_in[8];
    const float* e1 = (const float*)d_in[9];
    const float* W2 = (const float*)d_in[10];
    const float* b2 = (const float*)d_in[11];
    const float* g2 = (const float*)d_in[12];
    const float* e2 = (const float*)d_in[13];
    const float* W3 = (const float*)d_in[14];
    const float* b3 = (const float*)d_in[15];
    const float* g3 = (const float*)d_in[16];
    const float* e3 = (const float*)d_in[17];
    const float* W4 = (const float*)d_in[18];
    const float* b4 = (const float*)d_in[19];

    _Float16* wpack = (_Float16*)d_ws;

    // Pack weights (f32 -> f16, K padded to multiples of 32).
    pack_weights_kernel<<<( 32 *  32 + 255) / 256, 256, 0, stream>>>(W0, wpack + WOFF0,  32,   3,  32);
    pack_weights_kernel<<<( 64 *  32 + 255) / 256, 256, 0, stream>>>(W1, wpack + WOFF1,  64,  32,  32);
    pack_weights_kernel<<<(128 *  64 + 255) / 256, 256, 0, stream>>>(W2, wpack + WOFF2, 128,  64,  64);
    pack_weights_kernel<<<(256 * 128 + 255) / 256, 256, 0, stream>>>(W3, wpack + WOFF3, 256, 128, 128);
    pack_weights_kernel<<<(256 * 256 + 255) / 256, 256, 0, stream>>>(W4, wpack + WOFF4, 256, 256, 256);

    Params p;
    p.kpts = kpts; p.scores = scores; p.w = wpack;
    p.bias[0] = b0; p.bias[1] = b1; p.bias[2] = b2; p.bias[3] = b3; p.bias[4] = b4;
    p.gamma[0] = g0; p.gamma[1] = g1; p.gamma[2] = g2; p.gamma[3] = g3;
    p.beta[0] = e0; p.beta[1] = e1; p.beta[2] = e2; p.beta[3] = e3;
    p.out = (float*)d_out;

    // LDS: xs 67584 + ys 133120 + red 2048 + wbuf 65536 = 268288 B (<= 320 KB/WGP)
    const size_t smemBytes = (size_t)MT * XS * 2 + (size_t)MT * YS * 4 + (size_t)MT * 16
                           + (size_t)WBUF_HALVES * 2;
    const int grid = (BB * NN) / MT;   // 2048 workgroups
    keypoint_encoder_kernel<<<grid, 256, smemBytes, stream>>>(p);
}
